// MiniBatchKMeans_420906795495
// MI455X (gfx1250) — matrix operand order repaired and checked by reading, not compiled, not run on hardware
//
#include <hip/hip_runtime.h>
#include <hip/hip_bf16.h>

// ---------------------------------------------------------------------------
// MiniBatch K-Means assignment step on gfx1250 (CDNA5, wave32, WMMA).
//   distances argmin via  score = ||c||^2 - 2 x.c  (||x||^2 constant per row)
//   cross term via bf16x3 split WMMA (hi*hi + hi*lo + lo*hi) ~ fp32 accuracy
// ---------------------------------------------------------------------------

typedef __attribute__((ext_vector_type(16))) __bf16          v16bf;
typedef __attribute__((ext_vector_type(16))) unsigned short  v16u;
typedef __attribute__((ext_vector_type(8)))  float           v8f;

#define N_ROWS   262144
#define DIM      64
#define K_CL     1024
#define ROWS_WG  256          // 16 waves * 16 rows
#define STAGE_C  128          // centroids staged in LDS per stage

__device__ __forceinline__ v8f wmma_bf16(v16bf a, v16bf b, v8f c) {
    // D = A(16x32 bf16) x B(32x16 bf16) + C(16x16 f32)
    return __builtin_amdgcn_wmma_f32_16x16x32_bf16(
        /*neg_a=*/false, a, /*neg_b=*/false, b,
        /*c_mod=*/(short)0, c, /*reuse_a=*/false, /*reuse_b=*/false);
}

// Round-to-nearest-even fp32 -> bf16 split: f ~= hi + lo
__device__ __forceinline__ void split_bf16(float f, unsigned short& h, unsigned short& l) {
    unsigned int u  = __builtin_bit_cast(unsigned int, f);
    unsigned int r  = u + 0x7FFFu + ((u >> 16) & 1u);
    h = (unsigned short)(r >> 16);
    float hf = __builtin_bit_cast(float, (unsigned int)h << 16);
    float lf = f - hf;
    unsigned int u2 = __builtin_bit_cast(unsigned int, lf);
    unsigned int r2 = u2 + 0x7FFFu + ((u2 >> 16) & 1u);
    l = (unsigned short)(r2 >> 16);
}

// --------------------------- precompute kernels -----------------------------

__global__ __launch_bounds__(256)
void kmeans_zero(int* __restrict__ counts, float* __restrict__ sums) {
    int i = blockIdx.x * 256 + threadIdx.x;          // 66560 elements total
    if (i < K_CL) counts[i] = 0;
    if (i < K_CL * DIM) sums[i] = 0.0f;
}

__global__ __launch_bounds__(256)
void kmeans_split(const float* __restrict__ centroids,
                  unsigned short* __restrict__ c_hi,
                  unsigned short* __restrict__ c_lo) {
    int i = blockIdx.x * 256 + threadIdx.x;          // K_CL*DIM threads
    unsigned short h, l;
    split_bf16(centroids[i], h, l);
    c_hi[i] = h;
    c_lo[i] = l;
}

__global__ __launch_bounds__(256)
void kmeans_csq(const float* __restrict__ centroids, float* __restrict__ c_sq) {
    int k = blockIdx.x * 256 + threadIdx.x;          // K_CL threads
    const float* cp = centroids + (size_t)k * DIM;
    float s = 0.0f;
#pragma unroll
    for (int d = 0; d < DIM; ++d) s += cp[d] * cp[d];
    c_sq[k] = s;
}

// ------------------------------ main kernel --------------------------------
// Block: 512 threads = 16 waves. Each wave owns 16 rows, scans all 1024
// centroids in 16-wide N-chunks using bf16x3 WMMA, keeps running argmin in
// the f32 C-matrix layout, then scatters counts/sums with global atomics.

__global__ __launch_bounds__(512)
void kmeans_main(const float* __restrict__ batch,
                 const unsigned short* __restrict__ c_hi,
                 const unsigned short* __restrict__ c_lo,
                 const float* __restrict__ c_sq,
                 int*   __restrict__ assign,
                 int*   __restrict__ counts,
                 float* __restrict__ sums) {
    __shared__ __align__(32) unsigned short lds_hi[STAGE_C * DIM];   // 16 KB
    __shared__ __align__(32) unsigned short lds_lo[STAGE_C * DIM];   // 16 KB
    __shared__ float lds_csq[STAGE_C];

    const int tid     = threadIdx.x;
    const int lane    = tid & 31;
    const int wave    = tid >> 5;
    const int halfSel = lane >> 4;     // lane half: selects K-subrange
    const int lcol    = lane & 15;     // N / M index within tile
    const int rowBase = blockIdx.x * ROWS_WG + wave * 16;
    const int myRow   = rowBase + lcol;

    // ---- A tile in registers, reused for all centroids ----
    // 16-bit A 16x32 layout: elem i -> K = halfSel*16 + i (contiguous, matches B).
    const float* rp = batch + (size_t)myRow * DIM;
    v16bf A_hi[2], A_lo[2];
#pragma unroll
    for (int j = 0; j < 2; ++j) {                    // K-chunks of 32 dims
        v16u ah, al;
#pragma unroll
        for (int i = 0; i < 8; ++i) {
            unsigned short h0, l0, h1, l1;
            split_bf16(rp[j * 32 + halfSel * 16 + i],     h0, l0);
            split_bf16(rp[j * 32 + halfSel * 16 + 8 + i], h1, l1);
            ah[i] = h0;  al[i] = l0;
            ah[8 + i] = h1;  al[8 + i] = l1;
        }
        A_hi[j] = __builtin_bit_cast(v16bf, ah);
        A_lo[j] = __builtin_bit_cast(v16bf, al);
    }

    float bestS[8];
    int   bestI[8];
#pragma unroll
    for (int v = 0; v < 8; ++v) { bestS[v] = 3.4e38f; bestI[v] = 0; }

    for (int stage = 0; stage < K_CL / STAGE_C; ++stage) {
        const int cb = stage * STAGE_C;
        __syncthreads();
        // Cooperative stage fill: 2x16KB via 128-bit copies.
        {
            const uint4* sh = (const uint4*)(c_hi + (size_t)cb * DIM);
            const uint4* sl = (const uint4*)(c_lo + (size_t)cb * DIM);
            uint4* dh = (uint4*)lds_hi;
            uint4* dl = (uint4*)lds_lo;
            for (int t = tid; t < STAGE_C * DIM / 8; t += 512) {
                dh[t] = sh[t];
                dl[t] = sl[t];
            }
            if (tid < STAGE_C) lds_csq[tid] = c_sq[cb + tid];
        }
        __syncthreads();

        for (int ch = 0; ch < STAGE_C / 16; ++ch) {
            const int clocal = ch * 16 + lcol;       // this lane's centroid col
            // B (32x16 bf16): lane holds column lcol, elems i -> K = j*32 +
            // halfSel*16 + i  => 32 contiguous bytes in the staged centroid.
            const int boff0 = clocal * DIM + halfSel * 16;
            v16bf Bh0 = __builtin_bit_cast(v16bf, *(const v16u*)&lds_hi[boff0]);
            v16bf Bh1 = __builtin_bit_cast(v16bf, *(const v16u*)&lds_hi[boff0 + 32]);
            v16bf Bl0 = __builtin_bit_cast(v16bf, *(const v16u*)&lds_lo[boff0]);
            v16bf Bl1 = __builtin_bit_cast(v16bf, *(const v16u*)&lds_lo[boff0 + 32]);

            v8f acc = {};
            acc = wmma_bf16(A_hi[0], Bh0, acc);      // hi*hi   (dims 0..31)
            acc = wmma_bf16(A_hi[1], Bh1, acc);      // hi*hi   (dims 32..63)
            acc = wmma_bf16(A_hi[0], Bl0, acc);      // hi*lo
            acc = wmma_bf16(A_hi[1], Bl1, acc);
            acc = wmma_bf16(A_lo[0], Bh0, acc);      // lo*hi
            acc = wmma_bf16(A_lo[1], Bh1, acc);

            const float csq = lds_csq[ch * 16 + lcol];
            const int   n   = cb + ch * 16 + lcol;
#pragma unroll
            for (int v = 0; v < 8; ++v) {
                float s = __builtin_fmaf(-2.0f, acc[v], csq);
                if (s < bestS[v]) { bestS[v] = s; bestI[v] = n; }
            }
        }
    }

    // ---- per-row argmin across the 16 N-lanes (xor butterfly, tie -> low idx)
#pragma unroll
    for (int v = 0; v < 8; ++v) {
#pragma unroll
        for (int off = 1; off < 16; off <<= 1) {
            float os = __shfl_xor(bestS[v], off, 32);
            int   oi = __shfl_xor(bestI[v], off, 32);
            if (os < bestS[v] || (os == bestS[v] && oi < bestI[v])) {
                bestS[v] = os; bestI[v] = oi;
            }
        }
    }

    // Row m result: VGPR m%8, lane half m/8 (all lanes of that half agree now).
    if (lcol == 0) {                                 // lanes 0 and 16
#pragma unroll
        for (int v = 0; v < 8; ++v) {
            const int row = rowBase + v + 8 * halfSel;
            assign[row] = bestI[v];
            atomicAdd(&counts[bestI[v]], 1);
        }
    }

    // ---- cluster sums: whole wave scatters each of its 16 rows (2 dims/lane)
#pragma unroll
    for (int m = 0; m < 16; ++m) {
        const int a = __shfl(bestI[m & 7], (m >> 3) << 4, 32);
        const float* xr = batch + (size_t)(rowBase + m) * DIM;
        atomicAdd(&sums[a * DIM + lane * 2],     xr[lane * 2]);
        atomicAdd(&sums[a * DIM + lane * 2 + 1], xr[lane * 2 + 1]);
    }
}

// ------------------------------- launcher ----------------------------------

extern "C" void kernel_launch(void* const* d_in, const int* in_sizes, int n_in,
                              void* d_out, int out_size, void* d_ws, size_t ws_size,
                              hipStream_t stream) {
    const float* batch     = (const float*)d_in[0];   // [262144, 64] f32
    const float* centroids = (const float*)d_in[1];   // [1024, 64]   f32

    // d_out layout: [assignments i32 x N][counts i32 x K][sums f32 x K*D]
    int*   assign = (int*)d_out;
    int*   counts = (int*)d_out + N_ROWS;
    float* sums   = (float*)((int*)d_out + N_ROWS + K_CL);

    // workspace: centroid bf16 hi/lo planes + ||c||^2
    unsigned short* c_hi = (unsigned short*)d_ws;
    unsigned short* c_lo = c_hi + (size_t)K_CL * DIM;
    float*          c_sq = (float*)(c_lo + (size_t)K_CL * DIM);

    kmeans_zero <<<(K_CL * DIM + 255) / 256, 256, 0, stream>>>(counts, sums);
    kmeans_split<<<(K_CL * DIM) / 256, 256, 0, stream>>>(centroids, c_hi, c_lo);
    kmeans_csq  <<<K_CL / 256, 256, 0, stream>>>(centroids, c_sq);
    kmeans_main <<<N_ROWS / ROWS_WG, 512, 0, stream>>>(batch, c_hi, c_lo, c_sq,
                                                       assign, counts, sums);
}
